// RNN_1279900254811
// MI455X (gfx1250) — matrix operand order repaired and checked by reading, not compiled, not run on hardware
//
#include <hip/hip_runtime.h>

// ---------------------------------------------------------------------------
// Types
// ---------------------------------------------------------------------------
typedef __attribute__((ext_vector_type(16))) __bf16          v16bf;
typedef __attribute__((ext_vector_type(8)))  float           v8f;
typedef __attribute__((ext_vector_type(16))) unsigned short  u16x16;
typedef __attribute__((ext_vector_type(8)))  unsigned short  u16x8;

// ---------------------------------------------------------------------------
// Helpers
// ---------------------------------------------------------------------------
__device__ __forceinline__ unsigned short f2bf(float f) {
  unsigned u = __builtin_bit_cast(unsigned, f);
  u += 0x7FFFu + ((u >> 16) & 1u);          // round-to-nearest-even
  return (unsigned short)(u >> 16);
}
__device__ __forceinline__ float bf2f(unsigned short h) {
  unsigned u = ((unsigned)h) << 16;
  return __builtin_bit_cast(float, u);
}
// tanh via hardware exp2/rcp (TRANS ops)
__device__ __forceinline__ float fast_tanh(float x) {
  float ax = __builtin_fabsf(x);
  float e  = __builtin_amdgcn_exp2f(ax * -2.885390081777927f);  // 2*log2(e)
  float r  = (1.0f - e) * __builtin_amdgcn_rcpf(1.0f + e);
  return __builtin_copysignf(r, x);
}
__device__ __forceinline__ v16bf mk16(u16x8 lo, u16x8 hi) {
  u16x16 t;
#pragma unroll
  for (int i = 0; i < 8; ++i) { t[i] = lo[i]; t[i + 8] = hi[i]; }
  return __builtin_bit_cast(v16bf, t);
}

// ---------------------------------------------------------------------------
// f32 -> bf16 conversion
// ---------------------------------------------------------------------------
__global__ void cvt_f32_bf16(const float* __restrict__ in,
                             unsigned short* __restrict__ out, int n) {
  int i = blockIdx.x * blockDim.x + threadIdx.x;
  if (i < n) out[i] = f2bf(in[i]);
}

// ---------------------------------------------------------------------------
// Swizzle Wh (512x512 f32) into bf16 B-fragment order:
//   WhF[((kc*32 + w)*32 + lane)*16 + j] = bf16( Wh[k][col] )
//   with k = 32*kc + 16*(lane>>4) + j ,  col = 16*w + (lane&15)
// so the recurrence reads each fragment as contiguous b128 loads.
// ---------------------------------------------------------------------------
__global__ void swz_wh(const float* __restrict__ Wh,
                       unsigned short* __restrict__ WhF) {
  int idx = blockIdx.x * blockDim.x + threadIdx.x;   // 16*32*32 = 16384
  int lane = idx & 31, w = (idx >> 5) & 31, kc = idx >> 10;
  int col = w * 16 + (lane & 15);
  int kbase = kc * 32 + ((lane >> 4) << 4);
#pragma unroll
  for (int j = 0; j < 16; ++j)
    WhF[(size_t)idx * 16 + j] = f2bf(Wh[(size_t)(kbase + j) * 512 + col]);
}

// ---------------------------------------------------------------------------
// Tiled WMMA GEMM:  C[M][Nn] = A[M][K] * B + bias   (unchanged from round 2)
// ---------------------------------------------------------------------------
template <bool A_F32, bool B_NK, bool OUT_BF16, bool C_SWZ, bool C_NT>
__global__ __launch_bounds__(256) void gemm_wmma(
    const void* __restrict__ Ag, const unsigned short* __restrict__ Bg,
    const float* __restrict__ bias, void* __restrict__ Cg,
    int M, int K, int Nn) {
  constexpr int BM = 128, BN = 128, BK = 32, BKP = BK + 8;
  __shared__ unsigned short As[BM * BKP];
  __shared__ unsigned short Bs[BN * BKP];

  const int tid  = threadIdx.x;
  const int lane = tid & 31;
  const int wid  = tid >> 5;
  const int mbase = blockIdx.y * BM;
  const int nbase = blockIdx.x * BN;
  const int m0 = (wid >> 2) * 64;
  const int n0 = (wid & 3) * 32;

  v8f acc[4][2];
#pragma unroll
  for (int a = 0; a < 4; ++a)
#pragma unroll
    for (int b = 0; b < 2; ++b)
#pragma unroll
      for (int i = 0; i < 8; ++i) acc[a][b][i] = 0.0f;

  for (int kb = 0; kb < K; kb += BK) {
    {
      const int r = tid >> 1, k0 = (tid & 1) * 16;
      if (A_F32) {
        const float* A32 = (const float*)Ag + (size_t)(mbase + r) * K + kb + k0;
#pragma unroll
        for (int i = 0; i < 16; ++i) As[r * BKP + k0 + i] = f2bf(A32[i]);
        if (kb + BK < K) __builtin_prefetch(A32 + BK, 0, 0);
      } else {
        const unsigned short* A16 =
            (const unsigned short*)Ag + (size_t)(mbase + r) * K + kb + k0;
        *(u16x8*)&As[r * BKP + k0]     = *(const u16x8*)A16;
        *(u16x8*)&As[r * BKP + k0 + 8] = *(const u16x8*)(A16 + 8);
        if (kb + BK < K) __builtin_prefetch(A16 + BK, 0, 0);
      }
    }
    if (B_NK) {
      const int nn = tid >> 1, k0 = (tid & 1) * 16;
      const unsigned short* Bp = Bg + (size_t)(nbase + nn) * K + kb + k0;
      *(u16x8*)&Bs[nn * BKP + k0]     = *(const u16x8*)Bp;
      *(u16x8*)&Bs[nn * BKP + k0 + 8] = *(const u16x8*)(Bp + 8);
    } else {
      const int nn = tid & 127, k0 = (tid >> 7) * 16;
#pragma unroll
      for (int i = 0; i < 16; ++i)
        Bs[nn * BKP + k0 + i] = Bg[(size_t)(kb + k0 + i) * Nn + nbase + nn];
    }
    __syncthreads();

    v16bf afr[4];
#pragma unroll
    for (int mt = 0; mt < 4; ++mt) {
      const int row = m0 + mt * 16 + (lane & 15);
      const int kk  = (lane >> 4) * 8;
      afr[mt] = mk16(*(const u16x8*)&As[row * BKP + kk],
                     *(const u16x8*)&As[row * BKP + kk + 16]);
    }
    v16bf bfr[2];
#pragma unroll
    for (int nt = 0; nt < 2; ++nt) {
      const int col = n0 + nt * 16 + (lane & 15);
      const int kk  = (lane >> 4) * 8;
      bfr[nt] = mk16(*(const u16x8*)&Bs[col * BKP + kk],
                     *(const u16x8*)&Bs[col * BKP + kk + 16]);
    }
#pragma unroll
    for (int mt = 0; mt < 4; ++mt)
#pragma unroll
      for (int nt = 0; nt < 2; ++nt)
        acc[mt][nt] = __builtin_amdgcn_wmma_f32_16x16x32_bf16(
            false, afr[mt], false, bfr[nt], (short)0, acc[mt][nt], false, false);
    __syncthreads();
  }

#pragma unroll
  for (int mt = 0; mt < 4; ++mt) {
#pragma unroll
    for (int nt = 0; nt < 2; ++nt) {
      const int col = nbase + n0 + nt * 16 + (lane & 15);
      const float bv = bias[col];
#pragma unroll
      for (int i = 0; i < 8; ++i) {
        const int row = mbase + m0 + mt * 16 + i + ((lane >> 4) << 3);
        float v = acc[mt][nt][i] + bv;
        if (C_SWZ) {
          // row = n*1024 + t ; store in recurrence C-fragment order
          const int t  = row & 1023, nb = row >> 10;
          const int tn = col >> 4,  nl = col & 15;
          const int tm = nb >> 4,   ml = nb & 15;
          const size_t addr =
              ((((size_t)t * 32 + tn) * 4 + tm) * 32 + ((ml >> 3) * 16 + nl)) * 8 +
              (ml & 7);
          ((unsigned short*)Cg)[addr] = f2bf(v);
        } else if (C_NT) {
          const size_t ro = (size_t)(row & 63) * 1024 + (size_t)(row >> 6);
          ((float*)Cg)[ro * Nn + col] = v;
        } else if (OUT_BF16) {
          ((unsigned short*)Cg)[(size_t)row * Nn + col] = f2bf(v);
        } else {
          ((float*)Cg)[(size_t)row * Nn + col] = v;
        }
      }
    }
  }
}

// ---------------------------------------------------------------------------
// Sequential recurrence: 1 workgroup, 32 waves, T=1024 steps.
// B (Wh) sourcing per 32-wide K-chunk kc:
//   kc 0..6  : LDS-resident fragments (224KB, staged once)
//   kc 7..15 : streamed from L2 (WhF is 512KB -> L2-hot), 3-deep pipeline
// No register-resident weights -> fits 128-VGPR/wave budget at 32 waves.
// ---------------------------------------------------------------------------
__global__ __launch_bounds__(1024) void rnn_recurrent(
    const unsigned short* __restrict__ WhF, const float* __restrict__ h0,
    const unsigned short* __restrict__ XWs, unsigned short* __restrict__ Hall) {
  constexpr int H = 512, HP = H + 8, Nb = 64, T = 1024;
  constexpr int LDS_KC = 7;
  __shared__ unsigned short hbuf[Nb * HP];            // 66,560 B
  __shared__ unsigned short whl[LDS_KC * 32 * 512];   // 229,376 B

  const int tid = threadIdx.x, lane = tid & 31, w = tid >> 5;
  const int colg = w * 16 + (lane & 15);

  // ---- init h from h0 ----
#pragma unroll
  for (int i = 0; i < (Nb * H) / 1024; ++i) {
    int idx = i * 1024 + tid;
    hbuf[(idx >> 9) * HP + (idx & 511)] = f2bf(h0[idx]);
  }
  // ---- stage LDS-resident Wh fragments ----
  for (int kc = 0; kc < LDS_KC; ++kc) {
    const unsigned short* src = WhF + ((size_t)kc * 32 + w) * 512 + lane * 16;
    unsigned short* dst = &whl[(kc * 32 + w) * 512 + lane * 16];
    *(u16x8*)dst       = *(const u16x8*)src;
    *(u16x8*)(dst + 8) = *(const u16x8*)(src + 8);
  }
  __syncthreads();

  // XW fragment base: [t][tn=w][tm][lane][8]
  auto xw_ptr = [&](int t, int mt) -> const u16x8* {
    return (const u16x8*)(XWs + ((((size_t)t * 32 + w) * 4 + mt) * 32 + lane) * 8);
  };
  // L2-resident Wh fragment load (contiguous 32B per lane)
  auto ldG = [&](int kc, u16x8& a, u16x8& b) {
    const unsigned short* p = WhF + ((size_t)kc * 32 + w) * 512 + lane * 16;
    a = *(const u16x8*)p; b = *(const u16x8*)(p + 8);
  };
  // A fragment from h in LDS
  auto ldA = [&](int kc, int mt) -> v16bf {
    const int row = mt * 16 + (lane & 15);
    const int kk  = kc * 32 + ((lane >> 4) << 4);
    return mk16(*(const u16x8*)&hbuf[row * HP + kk],
                *(const u16x8*)&hbuf[row * HP + kk + 8]);
  };

  // software-pipelined XW prefetch (1 step ahead)
  u16x8 preXW[4];
#pragma unroll
  for (int mt = 0; mt < 4; ++mt) preXW[mt] = *xw_ptr(0, mt);

  for (int t = 0; t < T; ++t) {
    // consume prefetched XW tile directly into acc, then issue t+1 loads
    v8f acc[4];
#pragma unroll
    for (int mt = 0; mt < 4; ++mt)
#pragma unroll
      for (int i = 0; i < 8; ++i) acc[mt][i] = bf2f(preXW[mt][i]);
    const int tnx = (t + 1 < T) ? t + 1 : t;
#pragma unroll
    for (int mt = 0; mt < 4; ++mt) preXW[mt] = *xw_ptr(tnx, mt);

    // prime 3-deep pipeline of L2 Wh fragments
    u16x8 ga[3], gb[3];
#pragma unroll
    for (int q = 0; q < 3; ++q) ldG(LDS_KC + q, ga[q], gb[q]);

    // h_{t-1} @ Wh : 64 WMMAs per wave
#pragma unroll
    for (int kc = 0; kc < 16; ++kc) {
      v16bf bf;
      if (kc < LDS_KC) {
        const unsigned short* p = &whl[(kc * 32 + w) * 512 + lane * 16];
        bf = mk16(*(const u16x8*)p, *(const u16x8*)(p + 8));
      } else {
        const int q = (kc - LDS_KC) % 3;
        bf = mk16(ga[q], gb[q]);
        if (kc + 3 < 16) ldG(kc + 3, ga[q], gb[q]);
      }
#pragma unroll
      for (int mt = 0; mt < 4; ++mt)
        acc[mt] = __builtin_amdgcn_wmma_f32_16x16x32_bf16(
            false, ldA(kc, mt), false, bf, (short)0, acc[mt], false, false);
    }
    __syncthreads();  // all waves finished reading h_{t-1}

    // tanh -> h_t into LDS and time-major global H_all
#pragma unroll
    for (int mt = 0; mt < 4; ++mt) {
#pragma unroll
      for (int i = 0; i < 8; ++i) {
        int m = mt * 16 + i + ((lane >> 4) << 3);
        unsigned short hb = f2bf(fast_tanh(acc[mt][i]));
        hbuf[m * HP + colg] = hb;
        Hall[((size_t)t * Nb + m) * H + colg] = hb;
      }
    }
    __syncthreads();  // h_t visible before next step's reads
  }
}

// ---------------------------------------------------------------------------
// Launch
// ---------------------------------------------------------------------------
extern "C" void kernel_launch(void* const* d_in, const int* in_sizes, int n_in,
                              void* d_out, int out_size, void* d_ws, size_t ws_size,
                              hipStream_t stream) {
  (void)in_sizes; (void)n_in; (void)out_size; (void)ws_size;

  const float* X    = (const float*)d_in[0];   // (N,T,D)
  const float* h0   = (const float*)d_in[1];   // (N,H)
  const float* Wx   = (const float*)d_in[2];   // (D,H)
  const float* Wh   = (const float*)d_in[3];   // (H,H)
  const float* bvec = (const float*)d_in[4];   // (H)
  const float* fc_w = (const float*)d_in[5];   // (O,H)
  const float* fc_b = (const float*)d_in[6];   // (O)
  float* Out = (float*)d_out;                  // (N,T,O)

  constexpr int    Nb = 64, T = 1024, D = 256, H = 512, O = 256;
  constexpr size_t MT = (size_t)Nb * T;        // 65536 rows

  // workspace: XW swizzled bf16 | H_all time-major bf16 | Wx bf16 | fc_w bf16 | WhF
  unsigned short* XWs  = (unsigned short*)d_ws;
  unsigned short* Hall = XWs + MT * H;
  unsigned short* Wxb  = Hall + MT * H;
  unsigned short* fcwb = Wxb + (size_t)D * H;
  unsigned short* WhF  = fcwb + (size_t)O * H;  // 16*32*32*16 = 262144 elems

  cvt_f32_bf16<<<(D * H + 255) / 256, 256, 0, stream>>>(Wx, Wxb, D * H);
  cvt_f32_bf16<<<(O * H + 255) / 256, 256, 0, stream>>>(fc_w, fcwb, O * H);
  swz_wh<<<64, 256, 0, stream>>>(Wh, WhF);

  // Phase A: XW = X @ Wx + b  (A f32 rows n*T+t, B [K][N], out swizzled bf16)
  dim3 g1(H / 128, (unsigned)(MT / 128));
  gemm_wmma<true, false, true, true, false><<<g1, 256, 0, stream>>>(
      (const void*)X, Wxb, bvec, (void*)XWs, (int)MT, D, H);

  // Phase B: sequential recurrence over T
  rnn_recurrent<<<1, 1024, 0, stream>>>(WhF, h0, XWs, Hall);

  // Phase C: Out = H_all @ fc_w^T + fc_b
  dim3 g2(O / 128, (unsigned)(MT / 128));
  gemm_wmma<false, true, false, false, true><<<g2, 256, 0, stream>>>(
      (const void*)Hall, fcwb, fc_b, (void*)Out, (int)MT, H, O);
}
